// MultiHeadAttention_70918499992239
// MI455X (gfx1250) — matrix operand order repaired and checked by reading, not compiled, not run on hardware
//
#include <hip/hip_runtime.h>
#include <hip/hip_bf16.h>

typedef __bf16 bf16;
typedef __attribute__((ext_vector_type(16))) __bf16 v16bf;
typedef __attribute__((ext_vector_type(8)))  __bf16 v8bf;
typedef __attribute__((ext_vector_type(8)))  float  v8f;

union FragU { v16bf v; v8bf h[2]; };

__device__ __forceinline__ v8f zero8() {
  v8f z = {0.f,0.f,0.f,0.f,0.f,0.f,0.f,0.f};
  return z;
}

__device__ __forceinline__ v8f wmma_bf16(v16bf a, v16bf b, v8f c) {
  return __builtin_amdgcn_wmma_f32_16x16x32_bf16(false, a, false, b, (short)0, c, false, false);
}

// 16B global -> LDS async copy (ASYNCcnt-tracked, no VGPR round trip).
__device__ __forceinline__ void async_copy_b128(const bf16* gsrc, bf16* ldst) {
  uint32_t loff = (uint32_t)(uintptr_t)ldst;   // low 32 bits of flat addr == LDS offset
  uint64_t gaddr = (uint64_t)(uintptr_t)gsrc;
  asm volatile("global_load_async_to_lds_b128 %0, %1, off"
               :: "v"(loff), "v"(gaddr) : "memory");
}
__device__ __forceinline__ void wait_async_le8() {
  asm volatile("s_wait_asynccnt 0x8" ::: "memory");
}

// ---- 16-lane butterfly reductions in pure VALU via DPP16 (no LDS round trip).
// xor1 = quad_perm(1,0,3,2)=0xB1; xor2 = quad_perm(2,3,0,1)=0x4E;
// once quads uniform: row_half_mirror(0x141) == xor4, row_mirror(0x140) == xor8.
#if __has_builtin(__builtin_amdgcn_update_dpp)
template<int CTRL>
__device__ __forceinline__ float dpp_mov(float x) {
  return __builtin_bit_cast(float, __builtin_amdgcn_update_dpp(
      0, __builtin_bit_cast(int, x), CTRL, 0xf, 0xf, true));
}
__device__ __forceinline__ float red16_max(float v) {
  v = fmaxf(v, dpp_mov<0xB1>(v));
  v = fmaxf(v, dpp_mov<0x4E>(v));
  v = fmaxf(v, dpp_mov<0x141>(v));
  v = fmaxf(v, dpp_mov<0x140>(v));
  return v;
}
__device__ __forceinline__ float red16_sum(float v) {
  v += dpp_mov<0xB1>(v);
  v += dpp_mov<0x4E>(v);
  v += dpp_mov<0x141>(v);
  v += dpp_mov<0x140>(v);
  return v;
}
#else
__device__ __forceinline__ float red16_max(float v) {
  for (int off = 8; off >= 1; off >>= 1) v = fmaxf(v, __shfl_xor(v, off, 32));
  return v;
}
__device__ __forceinline__ float red16_sum(float v) {
  for (int off = 8; off >= 1; off >>= 1) v += __shfl_xor(v, off, 32);
  return v;
}
#endif

// ---------------- constants ----------------
constexpr int Bsz = 2;
constexpr int S   = 2048;
constexpr int D   = 1024;
constexpr int H   = 16;
constexpr int DH  = 64;
constexpr int M_ROWS = Bsz * S;          // 4096
constexpr float SCALE = 0.03125f;        // 1024^-0.5 (folded into Q projection)

// ---------------- fp32 -> bf16 convert (linear) ----------------
__global__ void cvt_f32_bf16(const float* __restrict__ in, bf16* __restrict__ out, int n) {
  int i = blockIdx.x * blockDim.x + threadIdx.x;
  int stride = gridDim.x * blockDim.x;
  for (; i < n; i += stride) out[i] = (bf16)in[i];
}

// ---------------- fp32 -> bf16 convert + transpose (1024x1024) ----------------
__global__ __launch_bounds__(256)
void cvtT_f32_bf16(const float* __restrict__ in, bf16* __restrict__ out) {
  __shared__ bf16 t[32][33];
  const int bx = blockIdx.x * 32;  // n base
  const int by = blockIdx.y * 32;  // k base
  const int tx = threadIdx.x, ty = threadIdx.y;   // 32 x 8
#pragma unroll
  for (int i = 0; i < 32; i += 8)
    t[ty + i][tx] = (bf16)in[(by + ty + i) * D + bx + tx];
  __syncthreads();
#pragma unroll
  for (int i = 0; i < 32; i += 8)
    out[(bx + ty + i) * D + by + tx] = t[tx][ty + i];
}

// ---------------- GEMM: C[M,N] = A[M,K] * Bt[N,K]^T ----------------
// OUTMODE: 0 bf16 row-major, 1 f32 row-major, 2 bf16 V-transposed, 3 bf16 * SCALE
constexpr int BM = 128, BN = 128, BK = 32;

template<int OUTMODE>
__global__ __launch_bounds__(128)
void gemm128(const bf16* __restrict__ A, const bf16* __restrict__ Bt,
             bf16* __restrict__ Cb, float* __restrict__ Cf) {
  __shared__ bf16 As[2][BM * BK];
  __shared__ bf16 Bs[2][BN * BK];

  const int tid  = threadIdx.x;
  const int lane = tid & 31;
  const int wave = tid >> 5;
  const int wm   = wave >> 1;
  const int wn   = wave & 1;
  const int l15  = lane & 15;
  const int hi   = lane >> 4;
  const int rb   = blockIdx.y * BM;
  const int nb   = blockIdx.x * BN;

  v8f acc[4][4];
#pragma unroll
  for (int i = 0; i < 4; i++)
#pragma unroll
    for (int j = 0; j < 4; j++) acc[i][j] = zero8();

  auto stage = [&](int buf, int k0) {
#pragma unroll
    for (int i = 0; i < 4; i++) {
      int cid = i * 128 + tid;
      int row = cid >> 2;
      int c4  = cid & 3;
      async_copy_b128(&A[(rb + row) * D + k0 + c4 * 8], &As[buf][row * BK + c4 * 8]);
    }
#pragma unroll
    for (int i = 0; i < 4; i++) {
      int cid = i * 128 + tid;
      int row = cid >> 2;
      int c4  = cid & 3;
      async_copy_b128(&Bt[(nb + row) * D + k0 + c4 * 8], &Bs[buf][row * BK + c4 * 8]);
    }
  };

  stage(0, 0);                          // 8 async instr outstanding
  int cur = 0;
  for (int k0 = 0; k0 < D; k0 += BK) {
    int knext = (k0 + BK) & (D - 1);    // wrap on last iter (harmless)
    stage(cur ^ 1, knext);              // outstanding <= 16
    wait_async_le8();                   // oldest 8 (buffer `cur`) complete, in order
    __syncthreads();

    v16bf af[4], bfr[4];
#pragma unroll
    for (int mt = 0; mt < 4; mt++) {
      int row = wm * 64 + mt * 16 + l15;
      FragU u;
      u.h[0] = *(const v8bf*)&As[cur][row * BK + hi * 8];
      u.h[1] = *(const v8bf*)&As[cur][row * BK + 16 + hi * 8];
      af[mt] = u.v;
    }
#pragma unroll
    for (int nt = 0; nt < 4; nt++) {
      int col = wn * 64 + nt * 16 + l15;
      FragU u;
      u.h[0] = *(const v8bf*)&Bs[cur][col * BK + hi * 8];
      u.h[1] = *(const v8bf*)&Bs[cur][col * BK + 16 + hi * 8];
      bfr[nt] = u.v;
    }
    __syncthreads();                    // reads of `cur` done -> may be overwritten

#pragma unroll
    for (int mt = 0; mt < 4; mt++)
#pragma unroll
      for (int nt = 0; nt < 4; nt++)
        acc[mt][nt] = wmma_bf16(af[mt], bfr[nt], acc[mt][nt]);
    cur ^= 1;
  }

#pragma unroll
  for (int mt = 0; mt < 4; mt++) {
#pragma unroll
    for (int nt = 0; nt < 4; nt++) {
      int row0 = rb + wm * 64 + mt * 16 + hi * 8;
      int col  = nb + wn * 64 + nt * 16 + l15;
#pragma unroll
      for (int j = 0; j < 8; j++) {
        int m = row0 + j;
        if (OUTMODE == 1) {
          Cf[m * D + col] = acc[mt][nt][j];
        } else if (OUTMODE == 2) {
          // Vt[(b*1024 + h*64 + d)][s]
          Cb[(((m >> 11) << 10) + col) * S + (m & (S - 1))] = (bf16)acc[mt][nt][j];
        } else if (OUTMODE == 3) {
          Cb[m * D + col] = (bf16)(acc[mt][nt][j] * SCALE);
        } else {
          Cb[m * D + col] = (bf16)acc[mt][nt][j];
        }
      }
    }
  }
}

// ---------------- flash attention ----------------
// grid: (S/64, B*H). block: 128 threads (4 waves), wave w owns 16 q-rows.
// Q pre-scaled by 1/sqrt(1024); K row-major; V pre-transposed Vt[(b*16+h)*64+d][s].
// K/V 64x64 tiles staged per-WG via async-to-LDS, double buffered.
__global__ __launch_bounds__(128)
void flash_attn(const bf16* __restrict__ Q, const bf16* __restrict__ K,
                const bf16* __restrict__ VT, bf16* __restrict__ AO) {
  __shared__ bf16 Ks[2][64 * 64];       // [k_local][d], 8 KB each
  __shared__ bf16 Vs[2][64 * 64];       // [d][s_local], 8 KB each
  __shared__ bf16 Pbuf[4 * 16 * 64];    // per-wave P tile, 8 KB

  const int tid  = threadIdx.x;
  const int lane = tid & 31;
  const int wave = tid >> 5;
  const int l15  = lane & 15;
  const int hi   = lane >> 4;

  const int qb = blockIdx.x * 64;
  const int bh = blockIdx.y;
  const int b  = bh >> 4;
  const int h  = bh & 15;

  const bf16* qp  = Q  + (b * S) * D + h * DH;
  const bf16* kp  = K  + (b * S) * D + h * DH;
  const bf16* vtp = VT + (bh * DH) * S;
  bf16*      aop  = AO + (b * S) * D + h * DH;

  const int qrow0 = qb + wave * 16;

  // Q A-fragments (two K=32 chunks), direct from global
  v16bf qf[2];
#pragma unroll
  for (int c = 0; c < 2; c++) {
    int dk0 = c * 32;
    FragU u;
    u.h[0] = *(const v8bf*)&qp[(qrow0 + l15) * D + dk0 + hi * 8];
    u.h[1] = *(const v8bf*)&qp[(qrow0 + l15) * D + dk0 + 16 + hi * 8];
    qf[c] = u.v;
  }

  float m_run[8], l_run[8];
#pragma unroll
  for (int j = 0; j < 8; j++) { m_run[j] = -3.0e38f; l_run[j] = 0.f; }
  v8f o[4];
#pragma unroll
  for (int dt = 0; dt < 4; dt++) o[dt] = zero8();

  bf16* Pw = Pbuf + wave * 16 * 64;
  const int kend = qb + 64;

  auto stage = [&](int buf, int kb) {
#pragma unroll
    for (int i = 0; i < 4; i++) {       // K tile: 64 rows x 64 d
      int cid = i * 128 + tid;
      int row = cid >> 3;
      int c8  = cid & 7;
      async_copy_b128(&kp[(kb + row) * D + c8 * 8], &Ks[buf][row * 64 + c8 * 8]);
    }
#pragma unroll
    for (int i = 0; i < 4; i++) {       // V^T tile: 64 d-rows x 64 s
      int cid = i * 128 + tid;
      int row = cid >> 3;
      int c8  = cid & 7;
      async_copy_b128(&vtp[row * S + kb + c8 * 8], &Vs[buf][row * 64 + c8 * 8]);
    }
  };

  stage(0, 0);
  int cur = 0;
  for (int kb = 0; kb < kend; kb += 64) {
    __syncthreads();                    // all reads of buf cur^1 (prev iter) done
    int kbn = (kb + 64 < kend) ? kb + 64 : 0;
    stage(cur ^ 1, kbn);
    wait_async_le8();                   // buffer `cur` landed
    __syncthreads();                    // visible to all waves

    // scores: S[16 q][64 k] = Q * K^T  (fragments from LDS)
    v8f s[4];
#pragma unroll
    for (int nt = 0; nt < 4; nt++) s[nt] = zero8();
#pragma unroll
    for (int nt = 0; nt < 4; nt++) {
#pragma unroll
      for (int c = 0; c < 2; c++) {
        int dko = c * 32 + hi * 8;
        FragU u;
        u.h[0] = *(const v8bf*)&Ks[cur][(nt * 16 + l15) * 64 + dko];
        u.h[1] = *(const v8bf*)&Ks[cur][(nt * 16 + l15) * 64 + dko + 16];
        s[nt] = wmma_bf16(qf[c], u.v, s[nt]);
      }
    }

    // online softmax; causal mask only needed on the diagonal block
    const bool diag = (kb == qb);
    float corr[8];
#pragma unroll
    for (int j = 0; j < 8; j++) {
      float tmax = -3.0e38f;
      if (diag) {
        int qg = qrow0 + hi * 8 + j;
#pragma unroll
        for (int nt = 0; nt < 4; nt++) {
          float tv = s[nt][j];
          if (kb + nt * 16 + l15 > qg) tv = -1.0e38f;
          s[nt][j] = tv;
          tmax = fmaxf(tmax, tv);
        }
      } else {
#pragma unroll
        for (int nt = 0; nt < 4; nt++) tmax = fmaxf(tmax, s[nt][j]);
      }
      tmax = red16_max(tmax);
      float mnew = fmaxf(m_run[j], tmax);
      corr[j] = __expf(m_run[j] - mnew);
      m_run[j] = mnew;
      float rs = 0.f;
#pragma unroll
      for (int nt = 0; nt < 4; nt++) {
        float pv = __expf(s[nt][j] - mnew);
        s[nt][j] = pv;
        rs += pv;
      }
      rs = red16_sum(rs);
      l_run[j] = l_run[j] * corr[j] + rs;
    }

#pragma unroll
    for (int nt = 0; nt < 4; nt++)
#pragma unroll
      for (int j = 0; j < 8; j++)
        Pw[(hi * 8 + j) * 64 + nt * 16 + l15] = (bf16)s[nt][j];
#pragma unroll
    for (int dt = 0; dt < 4; dt++)
#pragma unroll
      for (int j = 0; j < 8; j++) o[dt][j] *= corr[j];
    __syncthreads();                    // P visible; uniform trip count across waves

    // O += P * V  (both fragments from LDS)
#pragma unroll
    for (int c = 0; c < 2; c++) {
      FragU up;
      up.h[0] = *(const v8bf*)&Pw[l15 * 64 + c * 32 + hi * 8];
      up.h[1] = *(const v8bf*)&Pw[l15 * 64 + c * 32 + 16 + hi * 8];
#pragma unroll
      for (int dt = 0; dt < 4; dt++) {
        int col = dt * 16 + l15;
        FragU uv;
        uv.h[0] = *(const v8bf*)&Vs[cur][col * 64 + c * 32 + hi * 8];
        uv.h[1] = *(const v8bf*)&Vs[cur][col * 64 + c * 32 + 16 + hi * 8];
        o[dt] = wmma_bf16(up.v, uv.v, o[dt]);
      }
    }
    cur ^= 1;
  }

  // normalize and store attn-out (bf16, row-major)
#pragma unroll
  for (int dt = 0; dt < 4; dt++) {
#pragma unroll
    for (int j = 0; j < 8; j++) {
      int row = qrow0 + hi * 8 + j;
      aop[row * D + dt * 16 + l15] = (bf16)(o[dt][j] * (1.0f / l_run[j]));
    }
  }
}

// ---------------- launch ----------------
extern "C" void kernel_launch(void* const* d_in, const int* in_sizes, int n_in,
                              void* d_out, int out_size, void* d_ws, size_t ws_size,
                              hipStream_t stream) {
  const float* x  = (const float*)d_in[0];
  const float* Wq = (const float*)d_in[1];
  const float* Wk = (const float*)d_in[2];
  const float* Wv = (const float*)d_in[3];
  const float* Wo = (const float*)d_in[4];
  float* out = (float*)d_out;

  bf16* ws = (bf16*)d_ws;
  const int NX = M_ROWS * D;   // 4194304
  const int NW = D * D;        // 1048576
  bf16* xb  = ws;
  bf16* Wqt = ws + NX;
  bf16* Wkt = Wqt + NW;
  bf16* Wvt = Wkt + NW;
  bf16* Wot = Wvt + NW;
  bf16* Qb  = Wot + NW;
  bf16* Kb  = Qb + NX;
  bf16* VTb = Kb + NX;
  bf16* AOb = xb;                       // alias: x dead after QKV projections

  cvt_f32_bf16<<<2048, 256, 0, stream>>>(x, xb, NX);
  dim3 tb(32, 8), tg(32, 32);
  cvtT_f32_bf16<<<tg, tb, 0, stream>>>(Wq, Wqt);
  cvtT_f32_bf16<<<tg, tb, 0, stream>>>(Wk, Wkt);
  cvtT_f32_bf16<<<tg, tb, 0, stream>>>(Wv, Wvt);
  cvtT_f32_bf16<<<tg, tb, 0, stream>>>(Wo, Wot);

  dim3 gg(D / BN, M_ROWS / BM);         // (8, 32)
  gemm128<3><<<gg, 128, 0, stream>>>(xb, Wqt, Qb, nullptr);   // Q * (1/32)
  gemm128<0><<<gg, 128, 0, stream>>>(xb, Wkt, Kb, nullptr);
  gemm128<2><<<gg, 128, 0, stream>>>(xb, Wvt, VTb, nullptr);  // V transposed

  dim3 ga(S / 64, Bsz * H);             // (32, 32)
  flash_attn<<<ga, 128, 0, stream>>>(Qb, Kb, VTb, AOb);

  gemm128<1><<<gg, 128, 0, stream>>>(AOb, Wot, nullptr, out);
}
